// MEGNet_18726057411348
// MI455X (gfx1250) — compile-verified, matrix operand order checked by reading, not run on hardware
//
#include <hip/hip_runtime.h>

// ---------------------------------------------------------------------------
// MEGNet forward for MI455X (gfx1250, wave32, WMMA bf16).
// All edge/node MLPs run through v_wmma_f32_16x16x32_bf16 with f32 accum.
// Weights converted once per launch to bf16 transposed [N][Kpad] in d_ws.
// softplus2 uses native v_exp_f32/v_log_f32 (TRANS, co-executes with WMMA).
// ---------------------------------------------------------------------------

typedef __attribute__((ext_vector_type(16))) __bf16 v16bf;
typedef __attribute__((ext_vector_type(8)))  float  v8f;

union Frag16 { v16bf v; unsigned u[8]; };

__device__ __forceinline__ unsigned short f2bf(float x) {
    unsigned b = __float_as_uint(x);
    return (unsigned short)((b + 0x7FFFu + ((b >> 16) & 1u)) >> 16);
}

// softplus2(x) = log(exp(x)+1) - log(2), stable, via native exp2/log2:
//   max(x,0) + ln2*log2(1 + exp2(-|x|*log2e)) - ln2
__device__ __forceinline__ float sp2(float x) {
    float e = __builtin_amdgcn_exp2f(-fabsf(x) * 1.44269504f);   // v_exp_f32
    float l = __builtin_amdgcn_logf(1.0f + e);                   // v_log_f32 (log2)
    return fmaxf(x, 0.0f) + l * 0.69314718056f - 0.69314718056f;
}

// ---------------------------------------------------------------------------
// One MLP layer for one wave handling 16 rows of the 64-row tile.
//   actIn : LDS bf16 [64][ldi], K = padded input dim (mult of 32)
//   Wt    : LDS bf16 [N][K]  (transposed weights)
//   bias  : global f32 [N]
//   actOut: LDS bf16 [64][ldo], written at column offset co (or null)
//   fOut  : LDS f32  [64][ldf] (or null)
// A-frag: lane(l)->row r0+(l&15); VGPR v holds K pair
//         k = (v<4?0:16) + (l>>4)*8 + (v&3)*2  (ISA 16-bit 16x32 layout).
// B-frag mirrors A with the column index in the lane.
// ---------------------------------------------------------------------------
__device__ __forceinline__ void wave_layer(
    const unsigned short* actIn, int ldi,
    unsigned short* actOut, int ldo, int co,
    float* fOut, int ldf,
    const unsigned short* Wt, const float* bias,
    int K, int N, int r0, int lane)
{
    const int half = lane >> 4;
    const int lidx = lane & 15;
    for (int n0 = 0; n0 < N; n0 += 16) {
        v8f acc = {};
        const int n = n0 + lidx;
        for (int k0 = 0; k0 < K; k0 += 32) {
            Frag16 fa, fb;
            const unsigned short* ra = actIn + (r0 + lidx) * ldi + k0;
            const unsigned short* rb = Wt + n * K + k0;
#pragma unroll
            for (int v = 0; v < 8; ++v) {
                int kk = ((v < 4) ? 0 : 16) + half * 8 + (v & 3) * 2;
                fa.u[v] = *(const unsigned*)(ra + kk);
                fb.u[v] = *(const unsigned*)(rb + kk);
            }
            acc = __builtin_amdgcn_wmma_f32_16x16x32_bf16(
                false, fa.v, false, fb.v, (short)0, acc, false, false);
        }
        const float bv = bias[n];
#pragma unroll
        for (int i = 0; i < 8; ++i) {
            int m = r0 + half * 8 + i;     // D layout: VGPR i -> row i (+8 upper half)
            float y = sp2(acc[i] + bv);
            if (actOut) actOut[m * ldo + co + n] = f2bf(y);
            if (fOut)   fOut[m * ldf + n] = y;
        }
    }
}

// 16B-wide cooperative global->LDS copy (sizes are multiples of 16 bytes)
__device__ __forceinline__ void lds_copy(unsigned short* d, const unsigned short* s,
                                         int n, int tid, int nthr) {
    uint4* dd = (uint4*)d;
    const uint4* ss = (const uint4*)s;
    int n16 = n >> 3;
    for (int i = tid; i < n16; i += nthr) dd[i] = ss[i];
}

// ---------------------------------------------------------------------------
// Weight conversion: f32 [K][N] -> bf16 transposed [N][Kpad] (zero pad K)
// ---------------------------------------------------------------------------
__global__ void k_stage_w(const float* __restrict__ W, unsigned short* __restrict__ Wt,
                          int K, int N, int Kp) {
    int i = blockIdx.x * blockDim.x + threadIdx.x;
    if (i >= N * Kp) return;
    int n = i / Kp, k = i % Kp;
    Wt[i] = f2bf(k < K ? W[k * N + n] : 0.0f);
}

__global__ void k_count(const int* __restrict__ dst, float* cnt, int nE) {
    int i = blockIdx.x * blockDim.x + threadIdx.x;
    if (i < nE) atomicAdd(&cnt[dst[i]], 1.0f);
}
__global__ void k_invcnt(const float* cnt, float* inv, int n) {
    int i = blockIdx.x * blockDim.x + threadIdx.x;
    if (i < n) inv[i] = 1.0f / fmaxf(cnt[i], 1.0f);
}

// ---------------------------------------------------------------------------
// Fused Gaussian expansion + edge encoder: 100(pad128) -> 64 -> 32
// ---------------------------------------------------------------------------
__global__ __launch_bounds__(128) void k_gauss_encedge(
    const float* __restrict__ pos, const int* __restrict__ src, const int* __restrict__ dst,
    const unsigned short* __restrict__ W0t, const float* __restrict__ b0,
    const unsigned short* __restrict__ W1t, const float* __restrict__ b1,
    float* __restrict__ e_feat, int nE)
{
    __shared__ __align__(16) unsigned short actA[64 * 128];
    __shared__ __align__(16) unsigned short actB[64 * 64];
    __shared__ __align__(16) unsigned short wt[64 * 128];
    __shared__ float fOut[64 * 32];
    __shared__ float dist[64];
    int tid = threadIdx.x, lane = tid & 31, wave = tid >> 5;
    int base = blockIdx.x * 64;

    // speculative prefetch of next tile's index words (global_prefetch_b8)
    if (tid < 64) {
        int ne = base + 64 + tid;
        if (ne < nE) {
            __builtin_prefetch(&src[ne], 0, 1);
            __builtin_prefetch(&dst[ne], 0, 1);
        }
    }
    if (tid < 64) {
        int e = base + tid;
        float d = 0.0f;
        if (e < nE) {
            int s = src[e], t = dst[e];
            float dx = pos[t * 3 + 0] - pos[s * 3 + 0];
            float dy = pos[t * 3 + 1] - pos[s * 3 + 1];
            float dz = pos[t * 3 + 2] - pos[s * 3 + 2];
            d = sqrtf(dx * dx + dy * dy + dz * dz);
        }
        dist[tid] = d;
    }
    lds_copy(wt, W0t, 64 * 128, tid, 128);
    __syncthreads();
    for (int i = tid; i < 64 * 128; i += 128) {
        int r = i >> 7, c = i & 127;
        float v = 0.0f;
        if (c < 100) {
            float ctr = 4.0f * (float)c / 99.0f;
            float dd = dist[r] - ctr;
            v = __expf(-0.5f * dd * dd);
        }
        actA[i] = f2bf(v);
    }
    __syncthreads();
    wave_layer(actA, 128, actB, 64, 0, nullptr, 0, wt, b0, 128, 64, wave * 16, lane);
    __syncthreads();
    lds_copy(wt, W1t, 32 * 64, tid, 128);
    __syncthreads();
    wave_layer(actB, 64, nullptr, 0, 0, fOut, 32, wt, b1, 64, 32, wave * 16, lane);
    __syncthreads();
    for (int i = tid; i < 64 * 32; i += 128) {
        int r = i >> 5, c = i & 31, e = base + r;
        if (e < nE) e_feat[(size_t)e * 32 + c] = fOut[i];
    }
}

// ---------------------------------------------------------------------------
// Node encoder: embedding gather (16 pad 32) -> 64 -> 32
// ---------------------------------------------------------------------------
__global__ __launch_bounds__(128) void k_enc_node(
    const int* __restrict__ node_type, const float* __restrict__ emb,
    const unsigned short* __restrict__ W0t, const float* __restrict__ b0,
    const unsigned short* __restrict__ W1t, const float* __restrict__ b1,
    float* __restrict__ node_feat, int nN)
{
    __shared__ __align__(16) unsigned short actA[64 * 32];
    __shared__ __align__(16) unsigned short actB[64 * 64];
    __shared__ __align__(16) unsigned short wt[64 * 64];
    __shared__ float fOut[64 * 32];
    int tid = threadIdx.x, lane = tid & 31, wave = tid >> 5;
    int base = blockIdx.x * 64;

    for (int i = tid; i < 64 * 32; i += 128) {
        int r = i >> 5, c = i & 31, g = base + r;
        float v = 0.0f;
        if (g < nN && c < 16) v = emb[node_type[g] * 16 + c];
        actA[i] = f2bf(v);
    }
    lds_copy(wt, W0t, 64 * 32, tid, 128);
    __syncthreads();
    wave_layer(actA, 32, actB, 64, 0, nullptr, 0, wt, b0, 32, 64, wave * 16, lane);
    __syncthreads();
    lds_copy(wt, W1t, 32 * 64, tid, 128);
    __syncthreads();
    wave_layer(actB, 64, nullptr, 0, 0, fOut, 32, wt, b1, 64, 32, wave * 16, lane);
    __syncthreads();
    for (int i = tid; i < 64 * 32; i += 128) {
        int r = i >> 5, c = i & 31, g = base + r;
        if (g < nN) node_feat[(size_t)g * 32 + c] = fOut[i];
    }
}

// ---------------------------------------------------------------------------
// Generic row MLP 32 -> 64 -> 32 (pre-block node MLP)
// ---------------------------------------------------------------------------
__global__ __launch_bounds__(128) void k_mlp2_rows(
    const float* __restrict__ in,
    const unsigned short* __restrict__ W0t, const float* __restrict__ b0,
    const unsigned short* __restrict__ W1t, const float* __restrict__ b1,
    float* __restrict__ out, int nRows)
{
    __shared__ __align__(16) unsigned short actA[64 * 32];
    __shared__ __align__(16) unsigned short actB[64 * 64];
    __shared__ __align__(16) unsigned short wt[64 * 64];
    __shared__ float fOut[64 * 32];
    int tid = threadIdx.x, lane = tid & 31, wave = tid >> 5;
    int base = blockIdx.x * 64;

    for (int i = tid; i < 64 * 32; i += 128) {
        int r = i >> 5, c = i & 31, g = base + r;
        actA[i] = f2bf(g < nRows ? in[(size_t)g * 32 + c] : 0.0f);
    }
    lds_copy(wt, W0t, 64 * 32, tid, 128);
    __syncthreads();
    wave_layer(actA, 32, actB, 64, 0, nullptr, 0, wt, b0, 32, 64, wave * 16, lane);
    __syncthreads();
    lds_copy(wt, W1t, 32 * 64, tid, 128);
    __syncthreads();
    wave_layer(actB, 64, nullptr, 0, 0, fOut, 32, wt, b1, 64, 32, wave * 16, lane);
    __syncthreads();
    for (int i = tid; i < 64 * 32; i += 128) {
        int r = i >> 5, c = i & 31, g = base + r;
        if (g < nRows) out[(size_t)g * 32 + c] = fOut[i];
    }
}

// ---------------------------------------------------------------------------
// conv_e: (optional pre-edge 32->64->32) + gather concat [v_s,v_d,e',u']
//          -> 128 -> 64 -> 64 -> 32, epilogue: skip add in place,
//          ve scatter (segment sum), LDS-reduced graph sum.
// ---------------------------------------------------------------------------
__global__ __launch_bounds__(128) void k_conv_e(
    float* __restrict__ e_feat, const float* __restrict__ vprime,
    const float* __restrict__ uprime,
    const int* __restrict__ src, const int* __restrict__ dst,
    int hasPre,
    const unsigned short* __restrict__ PW0t, const float* __restrict__ Pb0,
    const unsigned short* __restrict__ PW1t, const float* __restrict__ Pb1,
    const unsigned short* __restrict__ W0t, const float* __restrict__ b0,
    const unsigned short* __restrict__ W1t, const float* __restrict__ b1,
    const unsigned short* __restrict__ W2t, const float* __restrict__ b2,
    float* __restrict__ ve, float* __restrict__ ue_sum, int nE)
{
    __shared__ __align__(16) unsigned short actC[64 * 128];
    __shared__ __align__(16) unsigned short actH[64 * 64];
    __shared__ __align__(16) unsigned short actP[64 * 64];
    __shared__ __align__(16) unsigned short actE[64 * 32];
    __shared__ __align__(16) unsigned short wt[64 * 128];
    __shared__ float fOut[64 * 32];
    __shared__ float ueAcc[32];
    int tid = threadIdx.x, lane = tid & 31, wave = tid >> 5;
    int base = blockIdx.x * 64;
    if (tid < 32) ueAcc[tid] = 0.0f;

    // speculative prefetch of next tile's streaming data
    if (tid < 64) {
        int ne = base + 64 + tid;
        if (ne < nE) {
            __builtin_prefetch(&e_feat[(size_t)ne * 32], 0, 1);
            __builtin_prefetch(&src[ne], 0, 1);
            __builtin_prefetch(&dst[ne], 0, 1);
        }
    }
    // load e0 tile (bf16) + node gathers + state broadcast into concat cols
    for (int i = tid; i < 64 * 32; i += 128) {
        int r = i >> 5, c = i & 31, e = base + r;
        float e0 = 0.0f, vs = 0.0f, vd = 0.0f;
        if (e < nE) {
            e0 = e_feat[(size_t)e * 32 + c];
            vs = vprime[(size_t)src[e] * 32 + c];
            vd = vprime[(size_t)dst[e] * 32 + c];
        }
        actE[i] = f2bf(e0);
        actC[r * 128 + c]      = f2bf(vs);
        actC[r * 128 + 32 + c] = f2bf(vd);
        actC[r * 128 + 96 + c] = f2bf(uprime[c]);
    }
    __syncthreads();
    if (hasPre) {
        lds_copy(wt, PW0t, 64 * 32, tid, 128);
        __syncthreads();
        wave_layer(actE, 32, actH, 64, 0, nullptr, 0, wt, Pb0, 32, 64, wave * 16, lane);
        __syncthreads();
        lds_copy(wt, PW1t, 32 * 64, tid, 128);
        __syncthreads();
        wave_layer(actH, 64, actC, 128, 64, nullptr, 0, wt, Pb1, 64, 32, wave * 16, lane);
        __syncthreads();
    } else {
        for (int i = tid; i < 64 * 32; i += 128)
            actC[(i >> 5) * 128 + 64 + (i & 31)] = actE[i];
        __syncthreads();
    }
    lds_copy(wt, W0t, 64 * 128, tid, 128);
    __syncthreads();
    wave_layer(actC, 128, actH, 64, 0, nullptr, 0, wt, b0, 128, 64, wave * 16, lane);
    __syncthreads();
    lds_copy(wt, W1t, 64 * 64, tid, 128);
    __syncthreads();
    wave_layer(actH, 64, actP, 64, 0, nullptr, 0, wt, b1, 64, 64, wave * 16, lane);
    __syncthreads();
    lds_copy(wt, W2t, 32 * 64, tid, 128);
    __syncthreads();
    wave_layer(actP, 64, nullptr, 0, 0, fOut, 32, wt, b2, 64, 32, wave * 16, lane);
    __syncthreads();
    // epilogue: skip add in place, segment-sum scatter, graph sum
    for (int i = tid; i < 64 * 32; i += 128) {
        int r = i >> 5, c = i & 31, e = base + r;
        if (e < nE) {
            float y = fOut[i];
            e_feat[(size_t)e * 32 + c] = y + e_feat[(size_t)e * 32 + c];
            atomicAdd(&ve[(size_t)dst[e] * 32 + c], y);
            atomicAdd(&ueAcc[c], y);
        }
    }
    __syncthreads();
    if (tid < 32) atomicAdd(&ue_sum[tid], ueAcc[tid]);
}

// ---------------------------------------------------------------------------
// conv_v: concat [v', ve*inv_cnt, u'] -> 96 -> 64 -> 64 -> 32, skip in place
// ---------------------------------------------------------------------------
__global__ __launch_bounds__(128) void k_conv_v(
    float* __restrict__ node_feat, const float* __restrict__ vprime,
    const float* __restrict__ ve, const float* __restrict__ inv_cnt,
    const float* __restrict__ uprime,
    const unsigned short* __restrict__ W0t, const float* __restrict__ b0,
    const unsigned short* __restrict__ W1t, const float* __restrict__ b1,
    const unsigned short* __restrict__ W2t, const float* __restrict__ b2,
    float* __restrict__ uv_sum, int nN)
{
    __shared__ __align__(16) unsigned short actC[64 * 96];
    __shared__ __align__(16) unsigned short actH[64 * 64];
    __shared__ __align__(16) unsigned short actP[64 * 64];
    __shared__ __align__(16) unsigned short wt[64 * 96];
    __shared__ float fOut[64 * 32];
    __shared__ float uvAcc[32];
    int tid = threadIdx.x, lane = tid & 31, wave = tid >> 5;
    int base = blockIdx.x * 64;
    if (tid < 32) uvAcc[tid] = 0.0f;

    for (int i = tid; i < 64 * 32; i += 128) {
        int r = i >> 5, c = i & 31, g = base + r;
        float a = 0.0f, m = 0.0f;
        if (g < nN) {
            a = vprime[(size_t)g * 32 + c];
            m = ve[(size_t)g * 32 + c] * inv_cnt[g];
        }
        actC[r * 96 + c]      = f2bf(a);
        actC[r * 96 + 32 + c] = f2bf(m);
        actC[r * 96 + 64 + c] = f2bf(uprime[c]);
    }
    lds_copy(wt, W0t, 64 * 96, tid, 128);
    __syncthreads();
    wave_layer(actC, 96, actH, 64, 0, nullptr, 0, wt, b0, 96, 64, wave * 16, lane);
    __syncthreads();
    lds_copy(wt, W1t, 64 * 64, tid, 128);
    __syncthreads();
    wave_layer(actH, 64, actP, 64, 0, nullptr, 0, wt, b1, 64, 64, wave * 16, lane);
    __syncthreads();
    lds_copy(wt, W2t, 32 * 64, tid, 128);
    __syncthreads();
    wave_layer(actP, 64, nullptr, 0, 0, fOut, 32, wt, b2, 64, 32, wave * 16, lane);
    __syncthreads();
    for (int i = tid; i < 64 * 32; i += 128) {
        int r = i >> 5, c = i & 31, g = base + r;
        if (g < nN) {
            float y = fOut[i];
            node_feat[(size_t)g * 32 + c] = y + node_feat[(size_t)g * 32 + c];
            atomicAdd(&uvAcc[c], y);
        }
    }
    __syncthreads();
    if (tid < 32) atomicAdd(&uv_sum[tid], uvAcc[tid]);
}

// ---------------------------------------------------------------------------
// Scalar small MLPs (state path, readout) — single row, VALU is fine.
// ---------------------------------------------------------------------------
__global__ __launch_bounds__(64) void k_state_mlp2(
    const float* __restrict__ in, int K,
    const float* __restrict__ W0, const float* __restrict__ b0, int N0,
    const float* __restrict__ W1, const float* __restrict__ b1, int N1,
    float* __restrict__ out)
{
    __shared__ float xin[96], h0[64];
    int t = threadIdx.x;
    for (int i = t; i < K; i += 64) xin[i] = in[i];
    __syncthreads();
    if (t < N0) {
        float a = b0[t];
        for (int k = 0; k < K; ++k) a += xin[k] * W0[k * N0 + t];
        h0[t] = sp2(a);
    }
    __syncthreads();
    if (t < N1) {
        float a = b1[t];
        for (int k = 0; k < N0; ++k) a += h0[k] * W1[k * N1 + t];
        out[t] = sp2(a);
    }
}

__global__ __launch_bounds__(64) void k_conv_u(
    float* __restrict__ u_state, const float* __restrict__ uprime,
    const float* __restrict__ ue_sum, const float* __restrict__ uv_sum,
    float invE, float invN,
    const float* __restrict__ W0, const float* __restrict__ b0,
    const float* __restrict__ W1, const float* __restrict__ b1,
    const float* __restrict__ W2, const float* __restrict__ b2)
{
    __shared__ float xin[96], h0[64], h1[64], u0s[32];
    int t = threadIdx.x;
    if (t < 32) {
        xin[t]      = uprime[t];
        xin[32 + t] = ue_sum[t] * invE;
        xin[64 + t] = uv_sum[t] * invN;
        u0s[t]      = u_state[t];
    }
    __syncthreads();
    { float a = b0[t]; for (int k = 0; k < 96; ++k) a += xin[k] * W0[k * 64 + t]; h0[t] = sp2(a); }
    __syncthreads();
    { float a = b1[t]; for (int k = 0; k < 64; ++k) a += h0[k] * W1[k * 64 + t]; h1[t] = sp2(a); }
    __syncthreads();
    if (t < 32) {
        float a = b2[t];
        for (int k = 0; k < 64; ++k) a += h1[k] * W2[k * 32 + t];
        u_state[t] = sp2(a) + u0s[t];
    }
}

__global__ void k_colsum(const float* __restrict__ x, int rows, float* __restrict__ acc) {
    __shared__ float s[32];
    int tid = threadIdx.x;
    if (tid < 32) s[tid] = 0.0f;
    __syncthreads();
    int total = rows * 32;
    float local = 0.0f;
    for (int i = blockIdx.x * blockDim.x + tid; i < total; i += gridDim.x * blockDim.x)
        local += x[i];                       // stride % 32 == 0 -> fixed column per thread
    atomicAdd(&s[tid & 31], local);
    __syncthreads();
    if (tid < 32) atomicAdd(&acc[tid], s[tid]);
}

__global__ __launch_bounds__(96) void k_readout(
    const float* __restrict__ nodeSum, const float* __restrict__ edgeSum,
    const float* __restrict__ u_state,
    const float* __restrict__ W1, const float* __restrict__ b1,
    const float* __restrict__ W2, const float* __restrict__ b2,
    const float* __restrict__ Wo, const float* __restrict__ bo,
    float* __restrict__ out, float invN, float invE)
{
    __shared__ float vec[96], h1[32], h2[16];
    int t = threadIdx.x;
    if (t < 32) {
        vec[t]      = nodeSum[t] * invN;
        vec[32 + t] = edgeSum[t] * invE;
        vec[64 + t] = u_state[t];
    }
    __syncthreads();
    if (t < 32) { float a = b1[t]; for (int k = 0; k < 96; ++k) a += vec[k] * W1[k * 32 + t]; h1[t] = sp2(a); }
    __syncthreads();
    if (t < 16) { float a = b2[t]; for (int k = 0; k < 32; ++k) a += h1[k] * W2[k * 16 + t]; h2[t] = sp2(a); }
    __syncthreads();
    if (t == 0) {
        float a = bo[0];
        for (int k = 0; k < 16; ++k) a += h2[k] * Wo[k];
        out[0] = a;
    }
}

// ---------------------------------------------------------------------------
// Host driver. Param leaf order assumes jax pytree flattening (sorted dict
// keys, None dropped): d_in[5..] =
//   blocks[0]{conv_e(6),conv_u(6),conv_v(6)}, blocks[1]{...,pre e/u/v(12)},
//   blocks[2]{...}, enc_edge(4), enc_node(4), enc_state(4), fc1(2), fc2(2),
//   node_emb(1), out(2)   -> bases: blk 5/23/53, enc_edge 83, ... out 100.
// ---------------------------------------------------------------------------
extern "C" void kernel_launch(void* const* d_in, const int* in_sizes, int n_in,
                              void* d_out, int out_size, void* d_ws, size_t ws_size,
                              hipStream_t stream) {
    (void)in_sizes; (void)n_in; (void)out_size; (void)ws_size;
    const int NN = 50000, NE = 800000;
    const int* node_type = (const int*)d_in[0];
    const float* pos     = (const float*)d_in[1];
    const int* src       = (const int*)d_in[2];
    const int* dst       = (const int*)d_in[3];
    const float* state_attr = (const float*)d_in[4];
    auto F = [&](int i) { return (const float*)d_in[i]; };

    char* wsb = (char*)d_ws; size_t off = 0;
    auto alloc = [&](size_t b) -> char* {
        char* p = wsb + off; off = (off + b + 255) & ~(size_t)255; return p;
    };
    float* e_feat    = (float*)alloc((size_t)NE * 32 * 4);
    float* node_feat = (float*)alloc((size_t)NN * 32 * 4);
    float* pre_v     = (float*)alloc((size_t)NN * 32 * 4);
    float* ve        = (float*)alloc((size_t)NN * 32 * 4);
    float* cnt       = (float*)alloc((size_t)NN * 4);
    float* inv_cnt   = (float*)alloc((size_t)NN * 4);
    float* u_state   = (float*)alloc(128);
    float* pre_u     = (float*)alloc(128);
    float* ue_sum    = (float*)alloc(128);
    float* uv_sum    = (float*)alloc(128);
    float* nodeSum   = (float*)alloc(128);
    float* edgeSum   = (float*)alloc(128);

    auto stage = [&](const float* W, int K, int N, int Kp) -> const unsigned short* {
        unsigned short* d = (unsigned short*)alloc((size_t)N * Kp * 2);
        int tot = N * Kp;
        k_stage_w<<<(tot + 255) / 256, 256, 0, stream>>>(W, d, K, N, Kp);
        return d;
    };

    // Stage all WMMA-path weights to bf16 transposed layout
    const unsigned short* ee0 = stage(F(83), 100, 64, 128);
    const unsigned short* ee1 = stage(F(85), 64, 32, 64);
    const unsigned short* en0 = stage(F(87), 16, 64, 32);
    const unsigned short* en1 = stage(F(89), 64, 32, 64);
    struct BlkW { const unsigned short *ce0,*ce1,*ce2,*cv0,*cv1,*cv2,*pe0,*pe1,*pv0,*pv1; };
    BlkW bw[3];
    const int bases[3] = {5, 23, 53};
    for (int b = 0; b < 3; ++b) {
        int B = bases[b];
        bw[b].ce0 = stage(F(B + 0), 128, 64, 128);
        bw[b].ce1 = stage(F(B + 2), 64, 64, 64);
        bw[b].ce2 = stage(F(B + 4), 64, 32, 64);
        bw[b].cv0 = stage(F(B + 12), 96, 64, 96);
        bw[b].cv1 = stage(F(B + 14), 64, 64, 64);
        bw[b].cv2 = stage(F(B + 16), 64, 32, 64);
        if (b > 0) {
            int Pb = B + 18;
            bw[b].pe0 = stage(F(Pb + 0), 32, 64, 32);
            bw[b].pe1 = stage(F(Pb + 2), 64, 32, 64);
            bw[b].pv0 = stage(F(Pb + 8), 32, 64, 32);
            bw[b].pv1 = stage(F(Pb + 10), 64, 32, 64);
        } else {
            bw[b].pe0 = bw[b].pe1 = bw[b].pv0 = bw[b].pv1 = nullptr;
        }
    }

    // Degree counts
    hipMemsetAsync(cnt, 0, (size_t)NN * 4, stream);
    k_count<<<(NE + 255) / 256, 256, 0, stream>>>(dst, cnt, NE);
    k_invcnt<<<(NN + 255) / 256, 256, 0, stream>>>(cnt, inv_cnt, NN);

    // Encoders
    k_gauss_encedge<<<NE / 64, 128, 0, stream>>>(pos, src, dst, ee0, F(84), ee1, F(86), e_feat, NE);
    k_enc_node<<<(NN + 63) / 64, 128, 0, stream>>>(node_type, F(99), en0, F(88), en1, F(90), node_feat, NN);
    k_state_mlp2<<<1, 64, 0, stream>>>(state_attr, 2, F(91), F(92), 64, F(93), F(94), 32, u_state);

    // 3 MEGNet blocks
    for (int b = 0; b < 3; ++b) {
        int B = bases[b], Pb = B + 18;
        const float* vprime = node_feat;
        const float* uprime = u_state;
        if (b > 0) {
            k_mlp2_rows<<<(NN + 63) / 64, 128, 0, stream>>>(
                node_feat, bw[b].pv0, F(Pb + 9), bw[b].pv1, F(Pb + 11), pre_v, NN);
            k_state_mlp2<<<1, 64, 0, stream>>>(u_state, 32, F(Pb + 4), F(Pb + 5), 64,
                                               F(Pb + 6), F(Pb + 7), 32, pre_u);
            vprime = pre_v; uprime = pre_u;
        }
        hipMemsetAsync(ve, 0, (size_t)NN * 32 * 4, stream);
        hipMemsetAsync(ue_sum, 0, 128, stream);
        hipMemsetAsync(uv_sum, 0, 128, stream);
        k_conv_e<<<NE / 64, 128, 0, stream>>>(
            e_feat, vprime, uprime, src, dst,
            b > 0 ? 1 : 0, bw[b].pe0, b > 0 ? F(Pb + 1) : nullptr,
            bw[b].pe1, b > 0 ? F(Pb + 3) : nullptr,
            bw[b].ce0, F(B + 1), bw[b].ce1, F(B + 3), bw[b].ce2, F(B + 5),
            ve, ue_sum, NE);
        k_conv_v<<<(NN + 63) / 64, 128, 0, stream>>>(
            node_feat, vprime, ve, inv_cnt, uprime,
            bw[b].cv0, F(B + 13), bw[b].cv1, F(B + 15), bw[b].cv2, F(B + 17),
            uv_sum, NN);
        k_conv_u<<<1, 64, 0, stream>>>(
            u_state, uprime, ue_sum, uv_sum, 1.0f / NE, 1.0f / NN,
            F(B + 6), F(B + 7), F(B + 8), F(B + 9), F(B + 10), F(B + 11));
    }

    // Readout
    hipMemsetAsync(nodeSum, 0, 128, stream);
    hipMemsetAsync(edgeSum, 0, 128, stream);
    k_colsum<<<512, 256, 0, stream>>>(node_feat, NN, nodeSum);
    k_colsum<<<512, 256, 0, stream>>>(e_feat, NE, edgeSum);
    k_readout<<<1, 96, 0, stream>>>(nodeSum, edgeSum, u_state,
                                    F(95), F(96), F(97), F(98), F(100), F(101),
                                    (float*)d_out, 1.0f / NN, 1.0f / NE);
}
